// ProtoNet_33964601376969
// MI455X (gfx1250) — matrix operand order repaired and checked by reading, not compiled, not run on hardware
//
#include <hip/hip_runtime.h>

typedef __attribute__((ext_vector_type(16))) _Float16 v16h;
typedef __attribute__((ext_vector_type(8)))  _Float16 v8h;
typedef __attribute__((ext_vector_type(8)))  float    v8f;

#define N_WORDS 262144
#define N_TAGS  512
#define BDIM    768
#define MDIM    128

union FragH {
    v16h v;
    v8h  h[2];
    _Float16 e[16];
};

__device__ __forceinline__ v8f zero8() {
    v8f z = {0.f, 0.f, 0.f, 0.f, 0.f, 0.f, 0.f, 0.f};
    return z;
}

__device__ __forceinline__ v8f wmma_f16(v16h a, v16h b, v8f c) {
    return __builtin_amdgcn_wmma_f32_16x16x32_f16(
        /*neg_a=*/false, a, /*neg_b=*/false, b,
        /*c_mod=*/(short)0, c, /*reuse_a=*/false, /*reuse_b=*/false);
}

// A-fragment (16x32 f16): lane-half holds contiguous 16-run.
// rowp = pointer to this lane's row; k0 already includes kbase + (lane>>4)*16.
__device__ __forceinline__ v16h fragA_h(const _Float16* rowp, int k0) {
    FragH f;
    f.h[0] = *(const v8h*)(rowp + k0);
    f.h[1] = *(const v8h*)(rowp + k0 + 8);
    return f.v;
}

// A-fragment built from f32 source with on-the-fly cvt.
__device__ __forceinline__ v16h fragA_f32(const float* rowp, int k0) {
    FragH f;
    const float4 a = *(const float4*)(rowp + k0);
    const float4 b = *(const float4*)(rowp + k0 + 4);
    const float4 c = *(const float4*)(rowp + k0 + 8);
    const float4 d = *(const float4*)(rowp + k0 + 12);
    f.e[0]  = (_Float16)a.x; f.e[1]  = (_Float16)a.y; f.e[2]  = (_Float16)a.z; f.e[3]  = (_Float16)a.w;
    f.e[4]  = (_Float16)b.x; f.e[5]  = (_Float16)b.y; f.e[6]  = (_Float16)b.z; f.e[7]  = (_Float16)b.w;
    f.e[8]  = (_Float16)c.x; f.e[9]  = (_Float16)c.y; f.e[10] = (_Float16)c.z; f.e[11] = (_Float16)c.w;
    f.e[12] = (_Float16)d.x; f.e[13] = (_Float16)d.y; f.e[14] = (_Float16)d.z; f.e[15] = (_Float16)d.w;
    return f.v;
}

// B-fragment (32x16 f16) from row-major B^T (NT GEMM): contiguous 16-run.
// k0 already includes kbase + (lane>>4)*16.
__device__ __forceinline__ v16h fragB_h(const _Float16* rowp, int k0) {
    FragH f;
    f.h[0] = *(const v8h*)(rowp + k0);
    f.h[1] = *(const v8h*)(rowp + k0 + 8);
    return f.v;
}

__device__ __forceinline__ v16h fragB_f32(const float* rowp, int k0) {
    FragH f;
    const float4 a = *(const float4*)(rowp + k0);
    const float4 b = *(const float4*)(rowp + k0 + 4);
    const float4 c = *(const float4*)(rowp + k0 + 8);
    const float4 d = *(const float4*)(rowp + k0 + 12);
    f.e[0]  = (_Float16)a.x; f.e[1]  = (_Float16)a.y; f.e[2]  = (_Float16)a.z; f.e[3]  = (_Float16)a.w;
    f.e[4]  = (_Float16)b.x; f.e[5]  = (_Float16)b.y; f.e[6]  = (_Float16)b.z; f.e[7]  = (_Float16)b.w;
    f.e[8]  = (_Float16)c.x; f.e[9]  = (_Float16)c.y; f.e[10] = (_Float16)c.z; f.e[11] = (_Float16)c.w;
    f.e[12] = (_Float16)d.x; f.e[13] = (_Float16)d.y; f.e[14] = (_Float16)d.z; f.e[15] = (_Float16)d.w;
    return f.v;
}

__device__ __forceinline__ float half16_max(float v) {
    v = fmaxf(v, __shfl_xor(v, 1, 32));
    v = fmaxf(v, __shfl_xor(v, 2, 32));
    v = fmaxf(v, __shfl_xor(v, 4, 32));
    v = fmaxf(v, __shfl_xor(v, 8, 32));
    return v;
}
__device__ __forceinline__ float half16_sum(float v) {
    v += __shfl_xor(v, 1, 32);
    v += __shfl_xor(v, 2, 32);
    v += __shfl_xor(v, 4, 32);
    v += __shfl_xor(v, 8, 32);
    return v;
}

// ---------------------------------------------------------------- kernel 0
__global__ void cvt_f32_to_f16_kernel(const float* __restrict__ in,
                                      _Float16* __restrict__ out, int n) {
    int i = blockIdx.x * 256 + threadIdx.x;
    if (i < n) out[i] = (_Float16)in[i];
}

// ---------------------------------------------------------------- kernel 1
// tags_m = tags_embed @ tags_W.T + tags_b ; also t_sq = rowwise ||tags_m||^2
// grid: 32 blocks x 256 threads. Block b handles tag rows 16b..16b+15;
// wave w handles metric cols 16w..16w+15 (one 16x16 WMMA tile, K-loop 768/32).
__global__ void __launch_bounds__(256)
tags_kernel(const float* __restrict__ tags_embed,
            const float* __restrict__ tags_W,
            const float* __restrict__ tags_b,
            _Float16* __restrict__ tags16,   // [512,128]
            float* __restrict__ t_sq) {      // [512]
    __shared__ float part[8][16];
    const int tid  = threadIdx.x;
    const int lane = tid & 31;
    const int wave = tid >> 5;
    const int l16  = lane & 15;
    const int kg   = lane >> 4;
    const int kgA  = kg * 16;
    const int kgB  = kg * 16;
    const int rowb = blockIdx.x * 16;

    const float* Arow = tags_embed + (size_t)(rowb + l16) * BDIM;
    const float* Brow = tags_W + (size_t)(wave * 16 + l16) * BDIM;

    v8f acc = zero8();
    for (int kb = 0; kb < BDIM; kb += 32) {
        v16h a = fragA_f32(Arow, kb + kgA);
        v16h b = fragB_f32(Brow, kb + kgB);
        acc = wmma_f16(a, b, acc);
    }

    const int col = wave * 16 + l16;
    const float bias = tags_b[col];
    float sq[8];
#pragma unroll
    for (int r = 0; r < 8; ++r) {
        float v = acc[r] + bias;
        tags16[(size_t)(rowb + r + kg * 8) * MDIM + col] = (_Float16)v;
        sq[r] = half16_sum(v * v);
    }
    if (l16 == 0) {
#pragma unroll
        for (int r = 0; r < 8; ++r) part[wave][r + kg * 8] = sq[r];
    }
    __syncthreads();
    if (tid < 16) {
        float tot = 0.f;
#pragma unroll
        for (int w = 0; w < 8; ++w) tot += part[w][tid];
        t_sq[rowb + tid] = tot;
    }
}

// ---------------------------------------------------------------- kernel 2
// Fused: words_m = words_embed@words_W.T + b (GEMM1, f32->f16 cvt on load),
// scores = words_m @ tags_m.T (GEMM2, from LDS-staged f16 words_m),
// logits = 2*scores - t_sq (row-shift-invariant => w_sq eliminated),
// then log_softmax + softmax written with nontemporal stores.
// grid: 4096 blocks x 256 threads; block = 64 word rows.
// wave w: row strip s = w>>1 (16 rows), tag-column half h = w&1 (256 cols).
__global__ void __launch_bounds__(256)
fused_kernel(const float* __restrict__ words_embed,   // [N,768] f32
             const _Float16* __restrict__ wW16,       // [128,768] f16
             const float* __restrict__ words_b,       // [128]
             const _Float16* __restrict__ tags16,     // [512,128] f16
             const float* __restrict__ t_sq,          // [512]
             float* __restrict__ out) {               // [2,N,512]
    __shared__ _Float16 wm16[64 * MDIM];  // 16 KB staged words_m
    __shared__ float redmax[2][64];
    __shared__ float redsum[2][64];

    const int tid  = threadIdx.x;
    const int lane = tid & 31;
    const int wave = tid >> 5;
    const int s    = wave >> 1;   // row strip 0..3
    const int h    = wave & 1;    // tag-col half 0..1
    const int l16  = lane & 15;
    const int kg   = lane >> 4;
    const int kgA  = kg * 16;
    const int kgB  = kg * 16;

    const size_t rowbase = (size_t)blockIdx.x * 64;

    // -------- GEMM1: words_m for 16 rows x 64 metric cols per wave --------
    const float* Arow = words_embed + (rowbase + s * 16 + l16) * BDIM;
    const int mbase = h * 64;
    v8f acc1[4];
#pragma unroll
    for (int t = 0; t < 4; ++t) acc1[t] = zero8();

    for (int kb = 0; kb < BDIM; kb += 32) {
        v16h a = fragA_f32(Arow, kb + kgA);
#pragma unroll
        for (int t = 0; t < 4; ++t) {
            v16h b = fragB_h(wW16 + (size_t)(mbase + t * 16 + l16) * BDIM, kb + kgB);
            acc1[t] = wmma_f16(a, b, acc1[t]);
        }
    }
    // bias + f16 stage to LDS
#pragma unroll
    for (int t = 0; t < 4; ++t) {
        const int col = mbase + t * 16 + l16;
        const float bias = words_b[col];
#pragma unroll
        for (int r = 0; r < 8; ++r) {
            const int lrow = s * 16 + r + kg * 8;
            wm16[lrow * MDIM + col] = (_Float16)(acc1[t][r] + bias);
        }
    }
    __syncthreads();

    // -------- GEMM2: scores 16 rows x 256 tag cols per wave (in regs) -----
    const _Float16* Alds = wm16 + (s * 16 + l16) * MDIM;
    const int tbase = h * 256;
    v8f acc[16];
#pragma unroll
    for (int t = 0; t < 16; ++t) acc[t] = zero8();

#pragma unroll
    for (int kb = 0; kb < MDIM; kb += 32) {
        v16h a = fragA_h(Alds, kb + kgA);   // ds_load_b128 from LDS
#pragma unroll
        for (int t = 0; t < 16; ++t) {
            v16h b = fragB_h(tags16 + (size_t)(tbase + t * 16 + l16) * MDIM, kb + kgB);
            acc[t] = wmma_f16(a, b, acc[t]);
        }
    }

    // logits = 2*score - t_sq[col]  (softmax is invariant to the w_sq shift)
#pragma unroll
    for (int t = 0; t < 16; ++t) {
        const float tsq = t_sq[tbase + t * 16 + l16];
#pragma unroll
        for (int r = 0; r < 8; ++r)
            acc[t][r] = fmaf(2.0f, acc[t][r], -tsq);
    }

    // ---- row max: in-registers over 16 tiles, then 16-lane shfl reduce ----
    float m[8];
#pragma unroll
    for (int r = 0; r < 8; ++r) {
        float v = acc[0][r];
#pragma unroll
        for (int t = 1; t < 16; ++t) v = fmaxf(v, acc[t][r]);
        m[r] = half16_max(v);
    }
    if (l16 == 0) {
#pragma unroll
        for (int r = 0; r < 8; ++r) redmax[h][s * 16 + r + kg * 8] = m[r];
    }
    __syncthreads();
    float gmax[8];
#pragma unroll
    for (int r = 0; r < 8; ++r) {
        const int lrow = s * 16 + r + kg * 8;
        gmax[r] = fmaxf(redmax[0][lrow], redmax[1][lrow]);
    }

    // ---- row sum of exp ----
    float sm[8];
#pragma unroll
    for (int r = 0; r < 8; ++r) {
        float v = 0.f;
#pragma unroll
        for (int t = 0; t < 16; ++t) v += __expf(acc[t][r] - gmax[r]);
        sm[r] = half16_sum(v);
    }
    if (l16 == 0) {
#pragma unroll
        for (int r = 0; r < 8; ++r) redsum[h][s * 16 + r + kg * 8] = sm[r];
    }
    __syncthreads();
    float lse[8], inv[8];
#pragma unroll
    for (int r = 0; r < 8; ++r) {
        const int lrow = s * 16 + r + kg * 8;
        const float gs = redsum[0][lrow] + redsum[1][lrow];
        lse[r] = __logf(gs);
        inv[r] = 1.0f / gs;
    }

    // ---- write outputs (streaming, never re-read -> nontemporal) ----
    const size_t PRED_OFF = (size_t)N_WORDS * N_TAGS;
#pragma unroll
    for (int t = 0; t < 16; ++t) {
        const int col = tbase + t * 16 + l16;
#pragma unroll
        for (int r = 0; r < 8; ++r) {
            const size_t row = rowbase + s * 16 + r + kg * 8;
            const size_t idx = row * N_TAGS + col;
            const float z = acc[t][r] - gmax[r];
            __builtin_nontemporal_store(z - lse[r], out + idx);
            __builtin_nontemporal_store(__expf(z) * inv[r], out + PRED_OFF + idx);
        }
    }
}

// ---------------------------------------------------------------- launcher
extern "C" void kernel_launch(void* const* d_in, const int* in_sizes, int n_in,
                              void* d_out, int out_size, void* d_ws, size_t ws_size,
                              hipStream_t stream) {
    const float* tags_embed  = (const float*)d_in[0];
    const float* words_embed = (const float*)d_in[1];
    const float* tags_W      = (const float*)d_in[2];
    const float* tags_b      = (const float*)d_in[3];
    const float* words_W     = (const float*)d_in[4];
    const float* words_b     = (const float*)d_in[5];
    float* out = (float*)d_out;

    char* ws = (char*)d_ws;
    _Float16* wW16   = (_Float16*)ws;                       // 128*768*2 = 196608 B
    _Float16* tags16 = (_Float16*)(ws + 196608);            // 512*128*2 = 131072 B
    float*    tsq    = (float*)(ws + 196608 + 131072);      // 512*4     =   2048 B

    cvt_f32_to_f16_kernel<<<(MDIM * BDIM + 255) / 256, 256, 0, stream>>>(
        words_W, wW16, MDIM * BDIM);
    tags_kernel<<<N_TAGS / 16, 256, 0, stream>>>(
        tags_embed, tags_W, tags_b, tags16, tsq);
    fused_kernel<<<N_WORDS / 64, 256, 0, stream>>>(
        words_embed, wW16, words_b, tags16, tsq, out);
}
